// Attention_52518860095570
// MI455X (gfx1250) — matrix operand order repaired and checked by reading, not compiled, not run on hardware
//
#include <hip/hip_runtime.h>
#include <hip/hip_bf16.h>
#include <stdint.h>

// ---- CDNA5 wave32 WMMA vector types -------------------------------------
typedef __attribute__((ext_vector_type(16))) __bf16         v16bf;
typedef __attribute__((ext_vector_type(8)))  float          v8f;
typedef __attribute__((ext_vector_type(16))) unsigned short v16u;
typedef __attribute__((ext_vector_type(8)))  unsigned short v8u;
typedef __attribute__((ext_vector_type(4)))  float          v4f;
// TDM descriptor group types (6-arg builtin signature on this toolchain)
typedef __attribute__((ext_vector_type(4))) unsigned int v4ui;
typedef __attribute__((ext_vector_type(8))) int          v8i;
typedef __attribute__((ext_vector_type(4))) int          v4i;

constexpr int kB = 2, kL = 2048, kD = 2048, kH = 16, kDh = 128;
constexpr int k3D = 3 * kD;

__device__ __forceinline__ unsigned short f2bf(float f) {
  unsigned int u = __float_as_uint(f);
  u += 0x7FFFu + ((u >> 16) & 1u);          // round-to-nearest-even
  return (unsigned short)(u >> 16);
}
__device__ __forceinline__ float bf2f(unsigned short h) {
  return __uint_as_float(((unsigned int)h) << 16);
}
// Assemble a 16xbf16 WMMA fragment from two contiguous 8-element halves.
__device__ __forceinline__ v16bf frag16(v8u lo, v8u hi) {
  v16u t;
#pragma unroll
  for (int i = 0; i < 8; ++i) { t[i] = lo[i]; t[i + 8] = hi[i]; }
  return __builtin_bit_cast(v16bf, t);
}

// ---- Tensor Data Mover: 2D tile (2-byte elements) global -> LDS ----------
// Builds D# group0/group1 per CDNA5 ISA §8.3/8.4; groups 2/3 zero (2D tile).
__device__ __forceinline__ void tdm_load_2d(
    unsigned lds_off, const void* gptr,
    unsigned tile_d0, unsigned tile_d1,            // elements per row / rows
    unsigned long long stride0_elems,              // row stride in elements
    unsigned pad_enable, unsigned pad_interval, unsigned pad_amount) {
  const unsigned long long ga = (unsigned long long)(uintptr_t)gptr;
  v4ui g0;
  g0[0] = 1u;                                      // count=1, user descriptor
  g0[1] = lds_off;                                 // lds_addr (bytes)
  g0[2] = (unsigned)(ga & 0xFFFFFFFFu);            // global_addr[31:0]
  g0[3] = (unsigned)((ga >> 32) & 0x01FFFFFFu) | (2u << 30);  // addr[56:32],type=2
  v8i g1;
  g1[0] = (int)((1u << 16) | (pad_enable << 20) |  // data_size=1 (2 bytes)
                (pad_interval << 22) | (pad_amount << 25));
  g1[1] = (int)((tile_d0 & 0xFFFFu) << 16);        // tensor_dim0 = tile_d0
  g1[2] = (int)(((tile_d0 >> 16) & 0xFFFFu) | ((tile_d1 & 0xFFFFu) << 16));
  g1[3] = (int)(((tile_d1 >> 16) & 0xFFFFu) | ((tile_d0 & 0xFFFFu) << 16)); // tile_dim0
  g1[4] = (int)(tile_d1 & 0xFFFFu);                // tile_dim1 (tile_dim2=0)
  g1[5] = (int)(stride0_elems & 0xFFFFFFFFu);      // tensor_dim0_stride[31:0]
  g1[6] = (int)((stride0_elems >> 32) & 0xFFFFu);  // stride[47:32]
  g1[7] = 0;
  const v4i gz4 = {};
  const v8i gz8 = {};
  __builtin_amdgcn_tensor_load_to_lds(g0, g1, gz4, gz4, gz8, 0);
}

// ---- fp32 -> bf16 cast, 8 elements / thread ------------------------------
__global__ __launch_bounds__(256) void cast_f32_to_bf16(
    const float* __restrict__ in, unsigned short* __restrict__ out, int n) {
  int i = (blockIdx.x * 256 + threadIdx.x) * 8;
  if (i + 8 <= n) {
    v4f a = *(const v4f*)(in + i);
    v4f b = *(const v4f*)(in + i + 4);
    v8u o;
#pragma unroll
    for (int j = 0; j < 4; ++j) { o[j] = f2bf(a[j]); o[j + 4] = f2bf(b[j]); }
    *(v8u*)(out + i) = o;
  }
}

// ---- bf16 WMMA GEMM: C[M,N] = A[M,K] @ B[K,N] ----------------------------
// 128x128 block tile, 8 waves, each wave 32x64 (2x4 16x16 frags), Kc = 32.
// A tile staged by the Tensor Data Mover (padded rows come from D# padding);
// B tile staged transposed by the waves.
template <int OUT_BF16>
__global__ __launch_bounds__(256) void gemm_bf16(
    const unsigned short* __restrict__ A, const unsigned short* __restrict__ Bm,
    void* __restrict__ Cout, int M, int N, int K) {
  __shared__ unsigned short As[128][40];   // [m][k], 16B pad per row (via TDM)
  __shared__ unsigned short Bs[128][40];   // transposed: [n][k], padded
  const int t = threadIdx.x;
  const int w = t >> 5, lane = t & 31;
  const int hf = lane >> 4, ml = lane & 15;
  const int wm = w & 3, wn = w >> 2;
  const int mblk = blockIdx.y, nblk = blockIdx.x;
  const unsigned as_off = (unsigned)(uintptr_t)(void*)&As[0][0];
  v8f acc[2][4];
  const v8f vz = {};
#pragma unroll
  for (int a = 0; a < 2; ++a)
#pragma unroll
    for (int b = 0; b < 4; ++b) acc[a][b] = vz;

  const int bk = t >> 3, bn0 = (t & 7) * 16;     // B: 32 rows x 128 cols
  for (int k0 = 0; k0 < K; k0 += 32) {
    // TDM: A[mblk*128 .. +127][k0 .. k0+31] -> As with 4-DWORD row padding
    if (w == 0)
      tdm_load_2d(as_off, A + (size_t)(mblk * 128) * K + k0,
                  /*tile_d0=*/32, /*tile_d1=*/128, (unsigned long long)K,
                  /*pad_enable=*/1, /*pad_interval=*/3, /*pad_amount=*/3);
    // waves stage B transposed
    const unsigned short* gb = Bm + (size_t)(k0 + bk) * N + nblk * 128 + bn0;
    v8u b0 = *(const v8u*)gb, b1 = *(const v8u*)(gb + 8);
#pragma unroll
    for (int j = 0; j < 8; ++j) { Bs[bn0 + j][bk] = b0[j]; Bs[bn0 + 8 + j][bk] = b1[j]; }
    if (k0 + 32 < K) __builtin_prefetch(gb + (size_t)32 * N, 0, 1);
    if (w == 0) __builtin_amdgcn_s_wait_tensorcnt(0);
    __syncthreads();

    v16bf af[2];
#pragma unroll
    for (int mf = 0; mf < 2; ++mf) {
      const int row = wm * 32 + mf * 16 + ml;    // A-frag: K = {0..7,16..23}+8*hf
      af[mf] = frag16(*(const v8u*)&As[row][hf * 8],
                      *(const v8u*)&As[row][16 + hf * 8]);
    }
#pragma unroll
    for (int nf = 0; nf < 4; ++nf) {
      const int col = wn * 64 + nf * 16 + ml;    // B-frag: K = {0..7,16..23}+8*hf
      v16bf bf = frag16(*(const v8u*)&Bs[col][hf * 8],
                        *(const v8u*)&Bs[col][16 + hf * 8]);
#pragma unroll
      for (int mf = 0; mf < 2; ++mf)
        acc[mf][nf] = __builtin_amdgcn_wmma_f32_16x16x32_bf16(
            false, af[mf], false, bf, (short)0, acc[mf][nf], false, false);
    }
    __syncthreads();
  }
#pragma unroll
  for (int mf = 0; mf < 2; ++mf)
#pragma unroll
    for (int nf = 0; nf < 4; ++nf)
#pragma unroll
      for (int r = 0; r < 8; ++r) {              // C layout: m = r + 8*hf, n = ml
        const int grow = mblk * 128 + wm * 32 + mf * 16 + r + hf * 8;
        const int gcol = nblk * 128 + wn * 64 + nf * 16 + ml;
        const float v = acc[mf][nf][r];
        if (OUT_BF16)
          ((unsigned short*)Cout)[(size_t)grow * N + gcol] = f2bf(v);
        else
          ((float*)Cout)[(size_t)grow * N + gcol] = v;
      }
}

// ---- RoPE + split into per-head [B,H,L,Dh] bf16 --------------------------
__global__ __launch_bounds__(256) void rope_split(
    const unsigned short* __restrict__ qkv, unsigned short* __restrict__ qo,
    unsigned short* __restrict__ ko, unsigned short* __restrict__ vo) {
  const int idx = blockIdx.x * 256 + threadIdx.x;   // kB*kL*kH*64 threads
  const int j = idx & 63;
  const int h = (idx >> 6) & (kH - 1);
  const int l = (idx >> 10) & (kL - 1);
  const int b = idx >> 21;
  if (b >= kB) return;
  const float invf = __expf(-(float)(2 * j) * (9.210340371976184f / 128.0f));
  float s, c;
  __sincosf((float)l * invf, &s, &c);
  const size_t base = ((size_t)(b * kL + l)) * k3D + h * kDh + j;
  const float q1 = bf2f(qkv[base]),      q2 = bf2f(qkv[base + 64]);
  const float k1 = bf2f(qkv[base + kD]), k2 = bf2f(qkv[base + kD + 64]);
  const size_t o = ((size_t)((b * kH + h) * kL + l)) * kDh + j;
  qo[o]      = f2bf(q1 * c - q2 * s);
  qo[o + 64] = f2bf(q2 * c + q1 * s);
  ko[o]      = f2bf(k1 * c - k2 * s);
  ko[o + 64] = f2bf(k2 * c + k1 * s);
  vo[o]      = qkv[base + 2 * kD];
  vo[o + 64] = qkv[base + 2 * kD + 64];
}

// ---- Flash attention: one WG per (b,h, 128-query block) ------------------
__global__ __launch_bounds__(256) void flash_attn(
    const unsigned short* __restrict__ Qm, const unsigned short* __restrict__ Km,
    const unsigned short* __restrict__ Vm, const int* __restrict__ amask,
    unsigned short* __restrict__ Y) {
  __shared__ unsigned short Ks[32][128];     // [key][dh] row-major (via TDM)
  __shared__ unsigned short Vs[128][40];     // transposed [dh][key], padded
  __shared__ float          Ss[8][16][36];   // per-wave scores 16x32, padded
  __shared__ unsigned short Ps[8][16][40];   // per-wave P (bf16), A-frag layout
  __shared__ float corrS[128], sumS[128];
  const int t = threadIdx.x;
  const int w = t >> 5, lane = t & 31;
  const int hf = lane >> 4, ml = lane & 15;
  const int qblk = blockIdx.x, h = blockIdx.y, b = blockIdx.z;
  const int qbase = qblk * 128;
  const size_t hoff = ((size_t)(b * kH + h)) * kL * kDh;
  const unsigned ks_off = (unsigned)(uintptr_t)(void*)&Ks[0][0];

  v16bf qf[4];                               // 16 rows x Dh=128 of Q in VGPRs
  {
    const unsigned short* qrowp = Qm + hoff + (size_t)(qbase + w * 16 + ml) * kDh;
#pragma unroll
    for (int kk = 0; kk < 4; ++kk) {
      const unsigned short* p = qrowp + kk * 32 + hf * 8;
      qf[kk] = frag16(*(const v8u*)p, *(const v8u*)(p + 16));
    }
  }
  v8f acc[8];
  const v8f vz = {};
#pragma unroll
  for (int i = 0; i < 8; ++i) acc[i] = vz;
  float mrun = -3.0e38f, lrun = 0.0f;        // lanes 0..15: row = w*16 + lane

  const int ck0 = t >> 3, cc0 = (t & 7) * 16;
  const int ktend = qbase + 128;             // causal upper bound (tile-aligned)
  for (int kt = 0; kt < ktend; kt += 32) {
    // TDM: K[kt..kt+31][0..127] -> Ks (row-major, no padding)
    if (w == 0)
      tdm_load_2d(ks_off, Km + hoff + (size_t)kt * kDh,
                  /*tile_d0=*/128, /*tile_d1=*/32, 128ull, 0, 0, 0);
    {  // stage V transposed (all waves)
      const unsigned short* gv = Vm + hoff + (size_t)(kt + ck0) * kDh + cc0;
      v8u vv0 = *(const v8u*)gv, vv1 = *(const v8u*)(gv + 8);
#pragma unroll
      for (int j = 0; j < 8; ++j) { Vs[cc0 + j][ck0] = vv0[j]; Vs[cc0 + 8 + j][ck0] = vv1[j]; }
    }
    if (w == 0) __builtin_amdgcn_s_wait_tensorcnt(0);
    __syncthreads();

    // S = Q @ K^T * scale, with causal + pad mask
#pragma unroll
    for (int nt = 0; nt < 2; ++nt) {
      v8f sa = vz;
#pragma unroll
      for (int kk = 0; kk < 4; ++kk) {
        const unsigned short* p = &Ks[nt * 16 + ml][kk * 32 + hf * 8];
        v16bf bf = frag16(*(const v8u*)p, *(const v8u*)(p + 16));
        sa = __builtin_amdgcn_wmma_f32_16x16x32_bf16(false, qf[kk], false, bf,
                                                     (short)0, sa, false, false);
      }
      const int kcol = kt + nt * 16 + ml;
      const int pmask = amask[b * kL + kcol];
#pragma unroll
      for (int r = 0; r < 8; ++r) {
        const int m = r + hf * 8;
        const int qrow = qbase + w * 16 + m;
        float sv = sa[r] * 0.08838834764831845f;      // 1/sqrt(128)
        if (kcol > qrow || pmask == 0) sv = -1.0e9f;
        Ss[w][m][nt * 16 + ml] = sv;
      }
    }
    __syncthreads();

    // online softmax: lane m owns row w*16+m
    if (lane < 16) {
      const int m = lane;
      float rmax = -3.0e38f;
#pragma unroll
      for (int c = 0; c < 32; ++c) rmax = fmaxf(rmax, Ss[w][m][c]);
      const float mnew = fmaxf(mrun, rmax);
      const float cr = __expf(mrun - mnew);
      float ps = 0.0f;
#pragma unroll
      for (int c = 0; c < 32; ++c) {
        const float p = __expf(Ss[w][m][c] - mnew);
        ps += p;
        Ps[w][m][c] = f2bf(p);
      }
      lrun = lrun * cr + ps;
      mrun = mnew;
      corrS[w * 16 + m] = cr;
    }
    __syncthreads();

    // rescale accumulators, then O += P @ V
#pragma unroll
    for (int r = 0; r < 8; ++r) {
      const float cr = corrS[w * 16 + r + hf * 8];
#pragma unroll
      for (int nn = 0; nn < 8; ++nn) acc[nn][r] *= cr;
    }
    v16bf pf = frag16(*(const v8u*)&Ps[w][ml][hf * 8],
                      *(const v8u*)&Ps[w][ml][16 + hf * 8]);
#pragma unroll
    for (int nn = 0; nn < 8; ++nn) {
      const unsigned short* pv = &Vs[nn * 16 + ml][hf * 8];
      v16bf vf = frag16(*(const v8u*)pv, *(const v8u*)(pv + 16));
      acc[nn] = __builtin_amdgcn_wmma_f32_16x16x32_bf16(false, pf, false, vf,
                                                        (short)0, acc[nn], false, false);
    }
    __syncthreads();
  }

  if (lane < 16) sumS[w * 16 + lane] = lrun;
  __syncthreads();
#pragma unroll
  for (int r = 0; r < 8; ++r) {
    const int m = r + hf * 8;
    const int qrow = qbase + w * 16 + m;
    const float inv = 1.0f / sumS[w * 16 + m];
    unsigned short* yp = Y + ((size_t)(b * kL + qrow)) * kD + (size_t)h * kDh;
#pragma unroll
    for (int nn = 0; nn < 8; ++nn) yp[nn * 16 + ml] = f2bf(acc[nn][r] * inv);
  }
}

// ---- host orchestration ---------------------------------------------------
extern "C" void kernel_launch(void* const* d_in, const int* in_sizes, int n_in,
                              void* d_out, int out_size, void* d_ws, size_t ws_size,
                              hipStream_t stream) {
  const float* x      = (const float*)d_in[0];
  const int*   amask  = (const int*)d_in[1];
  const float* w_qkv  = (const float*)d_in[2];
  const float* w_proj = (const float*)d_in[3];
  (void)in_sizes; (void)n_in; (void)out_size; (void)ws_size;

  char* ws = (char*)d_ws;
  size_t off = 0;
  auto take = [&](size_t elems) {
    unsigned short* p = (unsigned short*)(ws + off);
    off += ((elems * sizeof(unsigned short) + 255) & ~(size_t)255);
    return p;
  };
  unsigned short* xb     = take((size_t)kB * kL * kD);
  unsigned short* wqkvb  = take((size_t)kD * k3D);
  unsigned short* wprojb = take((size_t)kD * kD);
  unsigned short* qkvb   = take((size_t)kB * kL * k3D);
  unsigned short* qb     = take((size_t)kB * kH * kL * kDh);
  unsigned short* kb     = take((size_t)kB * kH * kL * kDh);
  unsigned short* vb     = take((size_t)kB * kH * kL * kDh);
  unsigned short* yb     = take((size_t)kB * kL * kD);

  cast_f32_to_bf16<<<(kB * kL * kD) / 2048, 256, 0, stream>>>(x, xb, kB * kL * kD);
  cast_f32_to_bf16<<<(kD * k3D) / 2048, 256, 0, stream>>>(w_qkv, wqkvb, kD * k3D);
  cast_f32_to_bf16<<<(kD * kD) / 2048, 256, 0, stream>>>(w_proj, wprojb, kD * kD);

  gemm_bf16<1><<<dim3(k3D / 128, (kB * kL) / 128), 256, 0, stream>>>(
      xb, wqkvb, qkvb, kB * kL, k3D, kD);

  rope_split<<<(kB * kL * kH * 64) / 256, 256, 0, stream>>>(qkvb, qb, kb, vb);

  flash_attn<<<dim3(kL / 128, kH, kB), 256, 0, stream>>>(qb, kb, vb, amask, yb);

  gemm_bf16<0><<<dim3(kD / 128, (kB * kL) / 128), 256, 0, stream>>>(
      yb, wprojb, d_out, kB * kL, kD, kD);
}
